// Mamba_Block_84722524881199
// MI455X (gfx1250) — matrix operand order repaired and checked by reading, not compiled, not run on hardware
//
#include <hip/hip_runtime.h>

typedef __attribute__((ext_vector_type(16))) _Float16 v16h;
typedef __attribute__((ext_vector_type(8)))  float    v8f;

#define BATCH   8
#define SEQ     4096
#define DMODEL  64
#define DSTATE  16
#define DCONV   4
#define DINNER  256
#define DTRANK  4
#define NROWS   (BATCH * SEQ)            /* 32768 token rows  */
#define NDBC    (DTRANK + 2 * DSTATE)    /* 36                */
#define CH      16                       /* scan chunks       */
#define CT      (SEQ / CH)               /* 256 steps / chunk */
#define TILE    64                       /* LDS staging tile  */

// ---------------------------------------------------------------------------
// CDNA5 async copy: global -> LDS directly (no VGPR bounce), ASYNCcnt-tracked.
// ldsoff = LDS byte offset (low 32 bits of generic shared pointer);
// gaddr  = 64-bit global address (GV mode).
// ---------------------------------------------------------------------------
__device__ __forceinline__ void async_ld_b32(void* lds_ptr, const float* gaddr) {
    unsigned ldsoff = (unsigned)(size_t)lds_ptr;
    asm volatile("global_load_async_to_lds_b32 %0, %1, off"
                 :: "v"(ldsoff), "v"(gaddr) : "memory");
}
__device__ __forceinline__ void async_wait0() {
    asm volatile("s_wait_asynccnt 0" ::: "memory");
}

// ---------------------------------------------------------------------------
// C[M x N] = A[M x K] * W[N x K]^T  (fp32 in/out, f16 WMMA compute, f32 acc)
// One wave computes NT adjacent 16x16 tiles (A fragment reused across NT
// B fragments -> NT v_wmma per K-step).  4 waves/block.
// A fragment (16-bit A 16x32): lane r=lane&15 -> row M=r, hh=lane>>4;
//   element e -> k = (e/8)*16 + hh*8 + (e%8)   => two contiguous float4 pairs
// B fragment (32x16): lane r -> col N=r; element e -> k = (e/8)*16 + hh*8 + (e%8)
//   => two contiguous float4 pairs
// C/D v8f: element i -> row = i + 8*hh, col = r
// ---------------------------------------------------------------------------
template <int NT>
__global__ void wmma_gemm_kernel(const float* __restrict__ A,
                                 const float* __restrict__ W,
                                 float* __restrict__ C,
                                 int K, int Ntot) {
    const int lane = threadIdx.x & 31;
    const int wave = threadIdx.x >> 5;
    const int m0   = blockIdx.x * 16;
    const int n0   = (blockIdx.y * 4 + wave) * (16 * NT);
    const int r    = lane & 15;
    const int hh   = lane >> 4;

    const float* __restrict__ arow = A + (size_t)(m0 + r) * K;
    const float* __restrict__ wrow = W + (size_t)(n0 + r) * K;

    v8f acc[NT];
#pragma unroll
    for (int tt = 0; tt < NT; ++tt)
        acc[tt] = (v8f){0.f, 0.f, 0.f, 0.f, 0.f, 0.f, 0.f, 0.f};

    for (int kb = 0; kb < K; kb += 32) {
        const float4* a0 = (const float4*)(arow + kb + (hh << 3));
        const float4* a1 = (const float4*)(arow + kb + 16 + (hh << 3));
        const float4 xa0 = a0[0], xa1 = a0[1], xa2 = a1[0], xa3 = a1[1];
        v16h av;
        av[0]=(_Float16)xa0.x; av[1]=(_Float16)xa0.y; av[2]=(_Float16)xa0.z; av[3]=(_Float16)xa0.w;
        av[4]=(_Float16)xa1.x; av[5]=(_Float16)xa1.y; av[6]=(_Float16)xa1.z; av[7]=(_Float16)xa1.w;
        av[8]=(_Float16)xa2.x; av[9]=(_Float16)xa2.y; av[10]=(_Float16)xa2.z; av[11]=(_Float16)xa2.w;
        av[12]=(_Float16)xa3.x; av[13]=(_Float16)xa3.y; av[14]=(_Float16)xa3.z; av[15]=(_Float16)xa3.w;

#pragma unroll
        for (int tt = 0; tt < NT; ++tt) {
            const float4* b0 = (const float4*)(wrow + (size_t)(tt * 16) * K + kb + (hh << 3));
            const float4* b1 = (const float4*)(wrow + (size_t)(tt * 16) * K + kb + 16 + (hh << 3));
            const float4 xb0 = b0[0], xb1 = b0[1], xb2 = b1[0], xb3 = b1[1];
            v16h bv;
            bv[0]=(_Float16)xb0.x; bv[1]=(_Float16)xb0.y; bv[2]=(_Float16)xb0.z; bv[3]=(_Float16)xb0.w;
            bv[4]=(_Float16)xb1.x; bv[5]=(_Float16)xb1.y; bv[6]=(_Float16)xb1.z; bv[7]=(_Float16)xb1.w;
            bv[8]=(_Float16)xb2.x; bv[9]=(_Float16)xb2.y; bv[10]=(_Float16)xb2.z; bv[11]=(_Float16)xb2.w;
            bv[12]=(_Float16)xb3.x; bv[13]=(_Float16)xb3.y; bv[14]=(_Float16)xb3.z; bv[15]=(_Float16)xb3.w;

            acc[tt] = __builtin_amdgcn_wmma_f32_16x16x32_f16(
                          false, av, false, bv, (short)0, acc[tt], false, false);
        }
    }
#pragma unroll
    for (int tt = 0; tt < NT; ++tt)
#pragma unroll
        for (int i = 0; i < 8; ++i) {
            const int row = m0 + i + (hh << 3);
            C[(size_t)row * Ntot + n0 + tt * 16 + r] = acc[tt][i];
        }
}

// ---------------------------------------------------------------------------
// Depthwise causal conv (width 4) + bias + SiLU.  xs = xz[..., :256]
// ---------------------------------------------------------------------------
__global__ void conv_silu_kernel(const float* __restrict__ xz,
                                 const float* __restrict__ conv_w,
                                 const float* __restrict__ conv_b,
                                 float* __restrict__ u) {
    const int idx = blockIdx.x * blockDim.x + threadIdx.x;  // (b*SEQ+l)*256 + d
    const int d   = idx & (DINNER - 1);
    const int bl  = idx >> 8;
    const int l   = bl & (SEQ - 1);
    float s = conv_b[d];
#pragma unroll
    for (int j = 0; j < DCONV; ++j) {
        const int ls = l - (DCONV - 1) + j;
        if (ls >= 0) {
            const float xv = xz[(size_t)(bl - (DCONV - 1) + j) * (2 * DINNER) + d];
            s += conv_w[d * DCONV + j] * xv;
        }
    }
    u[idx] = s / (1.f + __expf(-s));   // silu
}

// ---------------------------------------------------------------------------
// dbc[row][e] = sum_k u[row][k] * W_x[e][k]   (N=36, K=256; VALU dot)
// 288 threads/block = 8 rows x 36 outputs
// ---------------------------------------------------------------------------
__global__ void dbc_kernel(const float* __restrict__ u,
                           const float* __restrict__ W_x,
                           float* __restrict__ dbc) {
    const int t   = threadIdx.x;
    const int row = blockIdx.x * 8 + t / NDBC;
    const int e   = t % NDBC;
    const float* __restrict__ ur = u   + (size_t)row * DINNER;
    const float* __restrict__ wr = W_x + (size_t)e   * DINNER;
    float s = 0.f;
#pragma unroll 8
    for (int k = 0; k < DINNER; ++k) s += ur[k] * wr[k];
    dbc[(size_t)row * NDBC + e] = s;
}

// ---------------------------------------------------------------------------
// dt[row][d] = softplus( sum_r dbc[row][r] * W_dt[d][r] + b_dt[d] )
// ---------------------------------------------------------------------------
__global__ void dt_kernel(const float* __restrict__ dbc,
                          const float* __restrict__ W_dt,
                          const float* __restrict__ b_dt,
                          float* __restrict__ dt) {
    const int idx = blockIdx.x * blockDim.x + threadIdx.x;
    const int d   = idx & (DINNER - 1);
    const int row = idx >> 8;
    float s = b_dt[d];
#pragma unroll
    for (int r = 0; r < DTRANK; ++r)
        s += dbc[(size_t)row * NDBC + r] * W_dt[d * DTRANK + r];
    dt[idx] = (s > 20.f) ? s : log1pf(__expf(s));
}

// ---------------------------------------------------------------------------
// Chunked parallel scan, pass A: per chunk compute cumulative decay product
// P = prod exp(dt*A) and local end-state (h starting at 0).
// Block = 256 threads = 16 channels x 16 states, one (b, d-group, chunk).
// dt/u/dbc staged through LDS in 64-step tiles via async global->LDS copies.
// All staging loops have uniform trip counts -> EXEC stays all-ones.
// ---------------------------------------------------------------------------
__global__ void scan_chunk_kernel(const float* __restrict__ dt,
                                  const float* __restrict__ u,
                                  const float* __restrict__ dbc,
                                  const float* __restrict__ A_log,
                                  float* __restrict__ P,
                                  float* __restrict__ Hl) {
    __shared__ float s_dt[TILE][17];
    __shared__ float s_u[TILE][17];
    __shared__ float s_dbc[TILE][NDBC + 1];

    const int t  = threadIdx.x;
    const int n  = t & 15;
    const int ci = t >> 4;
    const int dg = blockIdx.x;
    const int c  = blockIdx.y;
    const int b  = blockIdx.z;
    const int d  = dg * 16 + ci;

    const float An = -__expf(A_log[d * DSTATE + n]);
    float h = 0.f, Pa = 1.f;

    for (int tile = 0; tile < CT / TILE; ++tile) {
        const int    l0   = c * CT + tile * TILE;
        const size_t row0 = (size_t)b * SEQ + l0;
        __syncthreads();                       // previous tile fully consumed
#pragma unroll
        for (int j = 0; j < TILE * 16; j += 256) {
            const int i = j + t, lo = i >> 4, doff = i & 15;
            async_ld_b32(&s_dt[lo][doff], dt + (row0 + lo) * DINNER + dg * 16 + doff);
            async_ld_b32(&s_u [lo][doff], u  + (row0 + lo) * DINNER + dg * 16 + doff);
        }
#pragma unroll
        for (int i = t; i < TILE * NDBC; i += 256)
            async_ld_b32(&s_dbc[i / NDBC][i % NDBC], dbc + row0 * NDBC + i);
        async_wait0();
        __syncthreads();
#pragma unroll 4
        for (int j = 0; j < TILE; ++j) {
            const float dtv = s_dt[j][ci];
            const float uv  = s_u [j][ci];
            const float Bv  = s_dbc[j][DTRANK + n];
            const float dA  = __expf(dtv * An);
            Pa *= dA;
            h = h * dA + dtv * uv * Bv;
        }
    }
    const size_t o = (((size_t)b * DINNER + d) * DSTATE + n) * CH + c;
    P [o] = Pa;
    Hl[o] = h;
}

// ---------------------------------------------------------------------------
// Pass B: sequential combine over CH chunks per (b,d,n) -> per-chunk init h.
// ---------------------------------------------------------------------------
__global__ void scan_combine_kernel(const float* __restrict__ P,
                                    const float* __restrict__ Hl,
                                    float* __restrict__ Hinit) {
    const int    idx  = blockIdx.x * blockDim.x + threadIdx.x;  // (b,d,n) flat
    const size_t base = (size_t)idx * CH;
    float h = 0.f;
#pragma unroll
    for (int c = 0; c < CH; ++c) {
        Hinit[base + c] = h;
        h = h * P[base + c] + Hl[base + c];
    }
}

// ---------------------------------------------------------------------------
// Pass C: re-run each chunk from its true init state; emit
// y = (sum_n h*C + u*D) * silu(z).  Reduction over n = 4 xor-shuffles.
// ---------------------------------------------------------------------------
__global__ void scan_emit_kernel(const float* __restrict__ dt,
                                 const float* __restrict__ u,
                                 const float* __restrict__ dbc,
                                 const float* __restrict__ xz,
                                 const float* __restrict__ A_log,
                                 const float* __restrict__ D_param,
                                 const float* __restrict__ Hinit,
                                 float* __restrict__ y) {
    __shared__ float s_dt[TILE][17];
    __shared__ float s_u[TILE][17];
    __shared__ float s_dbc[TILE][NDBC + 1];

    const int t  = threadIdx.x;
    const int n  = t & 15;
    const int ci = t >> 4;
    const int dg = blockIdx.x;
    const int c  = blockIdx.y;
    const int b  = blockIdx.z;
    const int d  = dg * 16 + ci;

    const float An = -__expf(A_log[d * DSTATE + n]);
    const float Dd = D_param[d];
    float h = Hinit[(((size_t)b * DINNER + d) * DSTATE + n) * CH + c];

    for (int tile = 0; tile < CT / TILE; ++tile) {
        const int    l0   = c * CT + tile * TILE;
        const size_t row0 = (size_t)b * SEQ + l0;
        if (tile + 1 < CT / TILE) {
            __builtin_prefetch(xz + (row0 + TILE) * (2 * DINNER) + DINNER + d, 0, 1);
        }
        __syncthreads();
#pragma unroll
        for (int j = 0; j < TILE * 16; j += 256) {
            const int i = j + t, lo = i >> 4, doff = i & 15;
            async_ld_b32(&s_dt[lo][doff], dt + (row0 + lo) * DINNER + dg * 16 + doff);
            async_ld_b32(&s_u [lo][doff], u  + (row0 + lo) * DINNER + dg * 16 + doff);
        }
#pragma unroll
        for (int i = t; i < TILE * NDBC; i += 256)
            async_ld_b32(&s_dbc[i / NDBC][i % NDBC], dbc + row0 * NDBC + i);
        async_wait0();
        __syncthreads();
        for (int j = 0; j < TILE; ++j) {
            const float dtv = s_dt[j][ci];
            const float uv  = s_u [j][ci];
            const float Bv  = s_dbc[j][DTRANK + n];
            const float Cv  = s_dbc[j][DTRANK + DSTATE + n];
            const float dA  = __expf(dtv * An);
            h = h * dA + dtv * uv * Bv;

            float p = h * Cv;
            p += __shfl_xor(p, 1);
            p += __shfl_xor(p, 2);
            p += __shfl_xor(p, 4);
            p += __shfl_xor(p, 8);

            if (n == 0) {
                const float zv = xz[(row0 + j) * (2 * DINNER) + DINNER + d];
                y[(row0 + j) * DINNER + d] =
                    (p + uv * Dd) * (zv / (1.f + __expf(-zv)));
            }
        }
    }
}

// ---------------------------------------------------------------------------
extern "C" void kernel_launch(void* const* d_in, const int* in_sizes, int n_in,
                              void* d_out, int out_size, void* d_ws, size_t ws_size,
                              hipStream_t stream) {
    const float* x     = (const float*)d_in[0];
    const float* W_in  = (const float*)d_in[1];
    const float* convw = (const float*)d_in[2];
    const float* convb = (const float*)d_in[3];
    const float* W_x   = (const float*)d_in[4];
    const float* W_dt  = (const float*)d_in[5];
    const float* b_dt  = (const float*)d_in[6];
    const float* A_log = (const float*)d_in[7];
    const float* D_par = (const float*)d_in[8];
    const float* W_out = (const float*)d_in[9];
    float* out = (float*)d_out;

    float* ws    = (float*)d_ws;
    float* xz    = ws;                                     // NROWS * 512
    float* u     = xz  + (size_t)NROWS * (2 * DINNER);     // NROWS * 256
    float* dbc   = u   + (size_t)NROWS * DINNER;           // NROWS * 36
    float* dtb   = dbc + (size_t)NROWS * NDBC;             // NROWS * 256
    float* y     = dtb + (size_t)NROWS * DINNER;           // NROWS * 256
    float* Pbuf  = y   + (size_t)NROWS * DINNER;           // B*DI*DS*CH
    float* Hl    = Pbuf + (size_t)BATCH * DINNER * DSTATE * CH;
    float* Hinit = Hl   + (size_t)BATCH * DINNER * DSTATE * CH;

    // 1) xz = x @ W_in^T   (M=32768, N=512, K=64): 2 N-tiles per wave
    wmma_gemm_kernel<2><<<dim3(NROWS / 16, (2 * DINNER) / 128), 128, 0, stream>>>(
        x, W_in, xz, DMODEL, 2 * DINNER);

    // 2) depthwise conv + SiLU -> u
    conv_silu_kernel<<<(NROWS * DINNER) / 256, 256, 0, stream>>>(xz, convw, convb, u);

    // 3) dbc = u @ W_x^T   (N=36)
    dbc_kernel<<<NROWS / 8, 288, 0, stream>>>(u, W_x, dbc);

    // 4) dt = softplus(dt_raw @ W_dt^T + b_dt)
    dt_kernel<<<(NROWS * DINNER) / 256, 256, 0, stream>>>(dbc, W_dt, b_dt, dtb);

    // 5a) chunked scan pass A: per-chunk decay product + local end state
    scan_chunk_kernel<<<dim3(DINNER / 16, CH, BATCH), 256, 0, stream>>>(
        dtb, u, dbc, A_log, Pbuf, Hl);

    // 5b) combine chunk states (sequential over CH per lane)
    scan_combine_kernel<<<(BATCH * DINNER * DSTATE) / 256, 256, 0, stream>>>(
        Pbuf, Hl, Hinit);

    // 5c) re-scan each chunk from true init; fuse +u*D and *silu(z) -> y
    scan_emit_kernel<<<dim3(DINNER / 16, CH, BATCH), 256, 0, stream>>>(
        dtb, u, dbc, xz, A_log, D_par, Hinit, y);

    // 6) out = y @ W_out^T  (M=32768, N=64, K=256)
    wmma_gemm_kernel<1><<<dim3(NROWS / 16, 1), 128, 0, stream>>>(
        y, W_out, out, DINNER, DMODEL);
}
